// Qwen3VLAttention_74259984548248
// MI455X (gfx1250) — matrix operand order repaired and checked by reading, not compiled, not run on hardware
//
#include <hip/hip_runtime.h>
#include <hip/hip_bf16.h>
#include <stdint.h>

typedef __bf16 bf16;
typedef bf16  v16bf __attribute__((ext_vector_type(16)));
typedef bf16  v8bf  __attribute__((ext_vector_type(8)));
typedef float v8f   __attribute__((ext_vector_type(8)));
typedef uint32_t u32x4 __attribute__((ext_vector_type(4)));

#define T_SEQ 2048
#define HIDN  4096
#define NQH   32
#define NKVH  8
#define DHD   128

#if __has_builtin(__builtin_amdgcn_tensor_load_to_lds)
#define HAVE_TDM 1
#else
#define HAVE_TDM 0
#endif

#if HAVE_TDM
typedef uint32_t tdm_u4 __attribute__((ext_vector_type(4)));
typedef int      tdm_i8 __attribute__((ext_vector_type(8)));
typedef int      tdm_i4 __attribute__((ext_vector_type(4)));

// Issue one 2D TDM tile load (bf16 elements) global -> LDS.
// D# packing per CDNA5 ISA.
// This toolchain's builtin takes 6 args: (g0 v4u, g1 v8i, g2 v4i, g3 v4i, v8i, cpol).
__device__ __forceinline__ void tdm_load_2d(const void* ldsp, const void* gptr,
                                            uint32_t tile_d0, uint32_t tile_d1,
                                            uint32_t tensor_d0, uint32_t tensor_d1,
                                            uint32_t stride0_elems) {
  uint64_t ga = (uint64_t)(uintptr_t)gptr;
  uint32_t lds_off = (uint32_t)(uintptr_t)ldsp;  // LDS aperture: low 32 bits = LDS offset
  tdm_u4 g0;
  g0[0] = 1u;                                   // count=1, user descriptor
  g0[1] = lds_off;                              // lds_addr
  g0[2] = (uint32_t)ga;                         // global_addr[31:0]
  g0[3] = (uint32_t)(ga >> 32) | (2u << 30);    // global_addr[56:32] | type=2 (image)
  tdm_i8 g1;
  g1[0] = (int)(1u << 16);                      // workgroup_mask=0 | data_size=1 (2 bytes)
  g1[1] = (int)((tensor_d0 & 0xFFFFu) << 16);   // atomic_barrier=0 | tensor_dim0[15:0]
  g1[2] = (int)((tensor_d0 >> 16) | ((tensor_d1 & 0xFFFFu) << 16));
  g1[3] = (int)((tensor_d1 >> 16) | (tile_d0 << 16));
  g1[4] = (int)tile_d1;                         // tile_dim1 | tile_dim2=0
  g1[5] = (int)stride0_elems;                   // tensor_dim0_stride[31:0]
  g1[6] = 0;
  g1[7] = 0;
  tdm_i4 z4 = {0, 0, 0, 0};
  tdm_i8 z8 = {0, 0, 0, 0, 0, 0, 0, 0};
  __builtin_amdgcn_tensor_load_to_lds(g0, g1, z4, z4, z8, 0);
}

__device__ __forceinline__ void tdm_wait0() {
#if __has_builtin(__builtin_amdgcn_s_wait_tensorcnt)
  __builtin_amdgcn_s_wait_tensorcnt(0);
#else
  asm volatile("s_wait_tensorcnt 0x0" ::: "memory");
#endif
}
__device__ __forceinline__ void tdm_wait1() {
#if __has_builtin(__builtin_amdgcn_s_wait_tensorcnt)
  __builtin_amdgcn_s_wait_tensorcnt(1);
#else
  asm volatile("s_wait_tensorcnt 0x1" ::: "memory");
#endif
}
__device__ __forceinline__ void tdm_wait2() {
#if __has_builtin(__builtin_amdgcn_s_wait_tensorcnt)
  __builtin_amdgcn_s_wait_tensorcnt(2);
#else
  asm volatile("s_wait_tensorcnt 0x2" ::: "memory");
#endif
}
#endif  // HAVE_TDM

// Load a 16-element bf16 fragment as two contiguous 16B vector loads (lo = e0..7, hi = e8..15).
__device__ __forceinline__ v16bf frag16(const bf16* p) {
  v8bf lo = *(const v8bf*)p;
  v8bf hi = *(const v8bf*)(p + 8);
  return __builtin_shufflevector(lo, hi, 0, 1, 2, 3, 4, 5, 6, 7, 8, 9, 10, 11, 12, 13, 14, 15);
}

// ---------------------------------------------------------------- cast f32 -> bf16
__global__ void cast_f32_bf16(const float* __restrict__ src, bf16* __restrict__ dst, int n) {
  int i = blockIdx.x * blockDim.x + threadIdx.x;
  if (i < n) dst[i] = (bf16)src[i];
}

// ---------------------------------------------------------------- transpose + cast: dst[n][k] = (bf16)src[k][n]
__global__ __launch_bounds__(256) void cast_transpose(const float* __restrict__ src,
                                                      bf16* __restrict__ dst,
                                                      int Kdim, int Ndim) {
  __shared__ float tile[32][33];
  const int nb2 = blockIdx.x * 32, kb = blockIdx.y * 32;
  const int tx = threadIdx.x & 31, ty = threadIdx.x >> 5;
#pragma unroll
  for (int i = ty; i < 32; i += 8)
    tile[i][tx] = src[(size_t)(kb + i) * Ndim + nb2 + tx];
  __syncthreads();
#pragma unroll
  for (int i = ty; i < 32; i += 8)
    dst[(size_t)(nb2 + i) * Kdim + kb + tx] = (bf16)tile[tx][i];
}

// ---------------------------------------------------------------- GEMM: C[M][N] = A[M][K] x Bt[N][K]^T, f32 out
// block: 256 thr = 8 waves as 2(m) x 4(n); block tile 32 x 256; wave tile 16 x 64 (4 accumulators).
// A staged in LDS (TDM double-buffered when available); B streamed from L2 (transposed, contiguous frags).
__global__ __launch_bounds__(256) void gemm_bf16_tn(const bf16* __restrict__ A,
                                                    const bf16* __restrict__ Bt,
                                                    float* __restrict__ C,
                                                    int M, int N, int K) {
  __shared__ __align__(16) bf16 As[2][32 * 32];
  const int tid = threadIdx.x, wave = tid >> 5, lane = tid & 31, ln = lane & 15;
  const int half8 = (lane < 16) ? 0 : 8, half16 = (lane < 16) ? 0 : 16;
  const int wm = wave >> 2, wn = wave & 3;
  const int m0 = blockIdx.y * 32;
  const int n0 = blockIdx.x * 256 + wn * 64;
  v8f acc[4] = {};
  const int nsteps = K >> 5;
#if HAVE_TDM
  if (wave == 0)
    tdm_load_2d(&As[0][0], A + (size_t)m0 * K, 32, 32,
                (uint32_t)K, (uint32_t)(M - m0), (uint32_t)K);
#endif
  for (int it = 0; it < nsteps; ++it) {
    const int k0 = it << 5;
    const int buf = it & 1;
#if HAVE_TDM
    if (wave == 0) {
      if (it + 1 < nsteps) {
        tdm_load_2d(&As[buf ^ 1][0], A + (size_t)m0 * K + (k0 + 32), 32, 32,
                    (uint32_t)(K - (k0 + 32)), (uint32_t)(M - m0), (uint32_t)K);
        tdm_wait1();
      } else {
        tdm_wait0();
      }
    }
#else
    {
      int i = tid * 4;
      int r = i >> 5, c = i & 31;
      *(uint64_t*)&As[buf][i] = *(const uint64_t*)&A[(size_t)(m0 + r) * K + k0 + c];
    }
#endif
    __syncthreads();
    v16bf a = frag16(&As[buf][(wm * 16 + ln) * 32 + half16]);
#pragma unroll
    for (int t = 0; t < 4; ++t) {
      const bf16* bp = Bt + (size_t)(n0 + t * 16 + ln) * K + k0 + half16;
      v16bf b = frag16(bp);
      acc[t] = __builtin_amdgcn_wmma_f32_16x16x32_bf16(false, a, false, b, (short)0, acc[t], false, false);
    }
    __syncthreads();
  }
  const int mrow = m0 + wm * 16 + half8;
#pragma unroll
  for (int t = 0; t < 4; ++t)
#pragma unroll
    for (int j = 0; j < 8; ++j)
      C[(size_t)(mrow + j) * N + n0 + t * 16 + ln] = acc[t][j];
}

// ---------------------------------------------------------------- fused per-head RMSNorm + mRoPE, f32 in -> bf16 [head][t][dh]
__global__ __launch_bounds__(128) void rmsnorm_rope(const float* __restrict__ inF,
                                                    const float* __restrict__ cosT,
                                                    const float* __restrict__ sinT,
                                                    const float* __restrict__ w,
                                                    bf16* __restrict__ outB,
                                                    int rowStride) {
  const int t = blockIdx.x, h = blockIdx.y, d = threadIdx.x;
  __shared__ float ssh[128];
  __shared__ float qsh[128];
  float v = inF[(size_t)t * rowStride + h * DHD + d];
  ssh[d] = v * v;
  __syncthreads();
  for (int s = 64; s > 0; s >>= 1) { if (d < s) ssh[d] += ssh[d + s]; __syncthreads(); }
  float inv = rsqrtf(ssh[0] * (1.0f / DHD) + 1e-6f);
  float nv = v * inv * w[d];
  qsh[d] = nv;
  __syncthreads();
  float rot = (d < 64) ? -qsh[d + 64] : qsh[d - 64];
  int dd = (d < 64) ? d : d - 64;
  int axis = (dd < 24) ? 0 : ((dd < 44) ? 1 : 2);
  size_t ci = ((size_t)axis * T_SEQ + t) * 64 + dd;
  float out = nv * cosT[ci] + rot * sinT[ci];
  outB[((size_t)h * T_SEQ + t) * DHD + d] = (bf16)out;
}

// ---------------------------------------------------------------- causal GQA flash attention
// grid: (T/128, NQH). block: 256 thr = 8 waves, each wave 16 q-rows.
// K stored [kv][t][d] (frag-contiguous), V stored transposed [kv][d][t] (frag-contiguous).
__global__ __launch_bounds__(256) void attn_fwd(const bf16* __restrict__ Qb,
                                                const bf16* __restrict__ Kb,
                                                const bf16* __restrict__ VbT,
                                                const float* __restrict__ amask,
                                                float* __restrict__ Ob) {
  const int qb = blockIdx.x, h = blockIdx.y, kv = h >> 2;
  const int tid = threadIdx.x, wave = tid >> 5, lane = tid & 31, ln = lane & 15;
  const int half8 = (lane < 16) ? 0 : 8, half16 = (lane < 16) ? 0 : 16;
  const int q0 = qb * 128 + wave * 16;
  __shared__ __align__(16) bf16 Kt[2][32 * DHD];   // [buf][key][d]
  __shared__ __align__(16) bf16 Vt[2][DHD * 32];   // [buf][d][key]
  __shared__ __align__(16) bf16 Psh[8][512];

  const bf16* Kbase = Kb + (size_t)kv * T_SEQ * DHD;   // [t][d]
  const bf16* Vbase = VbT + (size_t)kv * DHD * T_SEQ;  // [d][t]

  // preload Q fragments (16 rows x 128 dh = 4 A-fragments of 16x32)
  v16bf qfr[4];
  const bf16* Qrow = Qb + ((size_t)h * T_SEQ + q0 + ln) * DHD;
#pragma unroll
  for (int c = 0; c < 4; ++c) qfr[c] = frag16(Qrow + c * 32 + half16);

  v8f o[8] = {};
  float mrow[8], lrow[8];
#pragma unroll
  for (int j = 0; j < 8; ++j) { mrow[j] = -3.0e38f; lrow[j] = 0.0f; }
  const float scale = 0.08838834764831845f;  // 1/sqrt(128)
  const int nsteps = (qb * 128 + 128) >> 5;  // causal bound for this block

#if HAVE_TDM
  if (wave == 0) {
    tdm_load_2d(&Kt[0][0], Kbase, 128, 32, 128, (uint32_t)T_SEQ, 128);
    tdm_load_2d(&Vt[0][0], Vbase, 32, 128, (uint32_t)T_SEQ, 128, (uint32_t)T_SEQ);
  }
#endif
  for (int it = 0; it < nsteps; ++it) {
    const int k0 = it << 5;
    const int buf = it & 1;
#if HAVE_TDM
    if (wave == 0) {
      if (it + 1 < nsteps) {
        const int kn = k0 + 32;
        tdm_load_2d(&Kt[buf ^ 1][0], Kbase + (size_t)kn * DHD, 128, 32,
                    128, (uint32_t)(T_SEQ - kn), 128);
        tdm_load_2d(&Vt[buf ^ 1][0], Vbase + kn, 32, 128,
                    (uint32_t)(T_SEQ - kn), 128, (uint32_t)T_SEQ);
        tdm_wait2();  // leave the two just-issued outstanding; current pair complete
      } else {
        tdm_wait0();
      }
    }
#else
    {
      int e = tid * 16;
      { int r = e >> 7, c = e & 127;
        *(u32x4*)&Kt[buf][e] = *(const u32x4*)&Kbase[(size_t)(k0 + r) * DHD + c]; }
      { int e1 = e + 8; int r = e1 >> 7, c = e1 & 127;
        *(u32x4*)&Kt[buf][e1] = *(const u32x4*)&Kbase[(size_t)(k0 + r) * DHD + c]; }
      { int r = e >> 5, c = e & 31;
        *(u32x4*)&Vt[buf][e] = *(const u32x4*)&Vbase[(size_t)r * T_SEQ + k0 + c]; }
      { int e1 = e + 8; int r = e1 >> 5, c = e1 & 31;
        *(u32x4*)&Vt[buf][e1] = *(const u32x4*)&Vbase[(size_t)r * T_SEQ + k0 + c]; }
    }
#endif
    __syncthreads();

    if (k0 <= q0 + 15) {  // skip k-steps fully above this wave's causal bound
      // S = Q * K^T : two 16x16 tiles over 32 keys
      v8f sf[2];
#pragma unroll
      for (int kt = 0; kt < 2; ++kt) {
        v8f accv = {};
#pragma unroll
        for (int ch = 0; ch < 4; ++ch) {
          v16bf b = frag16(&Kt[buf][(kt * 16 + ln) * DHD + ch * 32 + half16]);
          accv = __builtin_amdgcn_wmma_f32_16x16x32_bf16(false, qfr[ch], false, b, (short)0, accv, false, false);
        }
        sf[kt] = accv;
      }
      // online softmax (row reductions across the 16-lane group via shfl_xor)
      float am0 = (1.0f - amask[k0 + ln]) * (-1.0e9f);
      float am1 = (1.0f - amask[k0 + 16 + ln]) * (-1.0e9f);
#pragma unroll
      for (int j = 0; j < 8; ++j) {
        const int m = q0 + half8 + j;
        float s0 = sf[0][j] * scale + am0 + ((k0 + ln)      > m ? -1.0e9f : 0.0f);
        float s1 = sf[1][j] * scale + am1 + ((k0 + 16 + ln) > m ? -1.0e9f : 0.0f);
        float rm = fmaxf(s0, s1);
#pragma unroll
        for (int off = 1; off < 16; off <<= 1) rm = fmaxf(rm, __shfl_xor(rm, off, 32));
        float mnew  = fmaxf(mrow[j], rm);
        float alpha = __expf(mrow[j] - mnew);
        mrow[j] = mnew;
        float p0 = __expf(s0 - mnew);
        float p1 = __expf(s1 - mnew);
        float ps = p0 + p1;
#pragma unroll
        for (int off = 1; off < 16; off <<= 1) ps += __shfl_xor(ps, off, 32);
        lrow[j] = lrow[j] * alpha + ps;
#pragma unroll
        for (int dt = 0; dt < 8; ++dt) o[dt][j] *= alpha;
        sf[0][j] = p0; sf[1][j] = p1;
      }
      // C-layout -> A-layout via per-wave LDS patch (bf16 P)
#pragma unroll
      for (int j = 0; j < 8; ++j) {
        Psh[wave][(half8 + j) * 32 + ln]      = (bf16)sf[0][j];
        Psh[wave][(half8 + j) * 32 + 16 + ln] = (bf16)sf[1][j];
      }
      v16bf pf = frag16(&Psh[wave][ln * 32 + half16]);
      // O += P * V  (8 output tiles across dh); V fragments contiguous from transposed tile
#pragma unroll
      for (int dt = 0; dt < 8; ++dt) {
        v16bf vfr = frag16(&Vt[buf][(dt * 16 + ln) * 32 + half16]);
        o[dt] = __builtin_amdgcn_wmma_f32_16x16x32_bf16(false, pf, false, vfr, (short)0, o[dt], false, false);
      }
    }
    __syncthreads();
  }
  // normalize + write [t][h*128 + d]
#pragma unroll
  for (int dt = 0; dt < 8; ++dt)
#pragma unroll
    for (int j = 0; j < 8; ++j) {
      float val = o[dt][j] / lrow[j];
      Ob[(size_t)(q0 + half8 + j) * HIDN + h * DHD + dt * 16 + ln] = val;
    }
}

// ---------------------------------------------------------------- host
extern "C" void kernel_launch(void* const* d_in, const int* in_sizes, int n_in,
                              void* d_out, int out_size, void* d_ws, size_t ws_size,
                              hipStream_t stream) {
  const float* x    = (const float*)d_in[0];
  const float* cosT = (const float*)d_in[1];
  const float* sinT = (const float*)d_in[2];
  const float* am   = (const float*)d_in[3];
  const float* Wq   = (const float*)d_in[4];
  const float* Wk   = (const float*)d_in[5];
  const float* Wv   = (const float*)d_in[6];
  const float* Wo   = (const float*)d_in[7];
  const float* qnw  = (const float*)d_in[8];
  const float* knw  = (const float*)d_in[9];
  float* out = (float*)d_out;

  char* ws = (char*)d_ws;
  size_t off = 0;
  auto alloc = [&](size_t bytes) -> void* {
    void* p = ws + off;
    off = (off + bytes + 255) & ~(size_t)255;
    return p;
  };
  const size_t NTH  = (size_t)T_SEQ * HIDN;
  const size_t NTKV = (size_t)T_SEQ * NKVH * DHD;

  bf16*  xb  = (bf16*) alloc(NTH * 2);
  bf16*  wqt = (bf16*) alloc((size_t)HIDN * HIDN * 2);        // Wq^T [n][k]
  bf16*  wkt = (bf16*) alloc((size_t)HIDN * NKVH * DHD * 2);  // Wk^T
  bf16*  wvt = (bf16*) alloc((size_t)HIDN * NKVH * DHD * 2);  // Wv^T
  bf16*  wot = (bf16*) alloc((size_t)HIDN * HIDN * 2);        // Wo^T
  float* qfw = (float*)alloc(NTH * 4);
  float* kfw = (float*)alloc(NTKV * 4);
  float* vfw = (float*)alloc(NTKV * 4);
  bf16*  qb2 = (bf16*) alloc(NTH * 2);    // [h][t][d]
  bf16*  kb2 = (bf16*) alloc(NTKV * 2);   // [kv][t][d]
  bf16*  vbt = (bf16*) alloc(NTKV * 2);   // [kv][d][t] (transposed)
  float* ofw = (float*)alloc(NTH * 4);
  bf16*  ob2 = (bf16*) alloc(NTH * 2);

  int n;
  n = (int)NTH; cast_f32_bf16<<<(n + 255) / 256, 256, 0, stream>>>(x, xb, n);
  cast_transpose<<<dim3(HIDN / 32, HIDN / 32), 256, 0, stream>>>(Wq, wqt, HIDN, HIDN);
  cast_transpose<<<dim3((NKVH * DHD) / 32, HIDN / 32), 256, 0, stream>>>(Wk, wkt, HIDN, NKVH * DHD);
  cast_transpose<<<dim3((NKVH * DHD) / 32, HIDN / 32), 256, 0, stream>>>(Wv, wvt, HIDN, NKVH * DHD);
  cast_transpose<<<dim3(HIDN / 32, HIDN / 32), 256, 0, stream>>>(Wo, wot, HIDN, HIDN);

  gemm_bf16_tn<<<dim3(HIDN / 256, T_SEQ / 32), 256, 0, stream>>>(xb, wqt, qfw, T_SEQ, HIDN, HIDN);
  gemm_bf16_tn<<<dim3((NKVH * DHD) / 256, T_SEQ / 32), 256, 0, stream>>>(xb, wkt, kfw, T_SEQ, NKVH * DHD, HIDN);
  gemm_bf16_tn<<<dim3((NKVH * DHD) / 256, T_SEQ / 32), 256, 0, stream>>>(xb, wvt, vfw, T_SEQ, NKVH * DHD, HIDN);

  rmsnorm_rope<<<dim3(T_SEQ, NQH),  128, 0, stream>>>(qfw, cosT, sinT, qnw, qb2, HIDN);
  rmsnorm_rope<<<dim3(T_SEQ, NKVH), 128, 0, stream>>>(kfw, cosT, sinT, knw, kb2, NKVH * DHD);
  // V: [t][kv*128+d] f32 -> [kv][d][t] bf16 == plain 2048x1024 transpose+cast
  cast_transpose<<<dim3((NKVH * DHD) / 32, T_SEQ / 32), 256, 0, stream>>>(vfw, vbt, T_SEQ, NKVH * DHD);

  attn_fwd<<<dim3(T_SEQ / 128, NQH), 256, 0, stream>>>(qb2, kb2, vbt, am, ofw);

  n = (int)NTH; cast_f32_bf16<<<(n + 255) / 256, 256, 0, stream>>>(ofw, ob2, n);
  gemm_bf16_tn<<<dim3(HIDN / 256, T_SEQ / 32), 256, 0, stream>>>(ob2, wot, out, T_SEQ, HIDN, HIDN);

  (void)in_sizes; (void)n_in; (void)out_size; (void)ws_size;
}
